// DualAttnUNet_8770323219127
// MI455X (gfx1250) — matrix operand order repaired and checked by reading, not compiled, not run on hardware
//
#include <hip/hip_runtime.h>
#include <hip/hip_bf16.h>
#include <stdint.h>

// ---------------------------------------------------------------------------
// DualAttnUNet block for MI455X (gfx1250, wave32, WMMA).
// HBM-bound (~0.5 GB traffic vs ~30 GFLOP) -> f32 activations in memory,
// bf16 conversion only in LDS staging, v_wmma_f32_16x16x32_bf16 everywhere.
// v2: instance-norm stats fused into GEMM epilogue (atomics + shfl reduce),
//     double-buffered LDS staging with software pipelining, K-loop unrolled.
// ---------------------------------------------------------------------------

#define CI    128
#define DVOL  32768     // 32*32*32
#define LLEN  128
#define CT    768
#define NH    4
#define HD    32
#define NB    2
#define EPSV  1e-5f

typedef __bf16 bf16_t;
typedef bf16_t v16bf __attribute__((ext_vector_type(16)));
typedef float  v8f   __attribute__((ext_vector_type(8)));

union BfPack { uint32_t u[8]; uint4 q[2]; v16bf v; };

__device__ __forceinline__ uint16_t f2bf(float f) {
  uint32_t x = __float_as_uint(f);
  x = (x + 0x7FFFu + ((x >> 16) & 1u)) >> 16;       // round-to-nearest-even
  return (uint16_t)x;
}
__device__ __forceinline__ uint32_t pack2(float a, float b) {
  return (uint32_t)f2bf(a) | ((uint32_t)f2bf(b) << 16);
}

// ---------------------------------------------------------------------------
// Pack the six 128x128 f32 weight matrices to bf16 pairs (done once, tiny).
// ---------------------------------------------------------------------------
__global__ void pack_weights_k(const float* __restrict__ w0, const float* __restrict__ w1,
                               const float* __restrict__ w2, const float* __restrict__ w3,
                               const float* __restrict__ w4, const float* __restrict__ w5,
                               uint32_t* __restrict__ out) {
  const float* ws[6] = {w0, w1, w2, w3, w4, w5};
  int tid = blockIdx.x * blockDim.x + threadIdx.x;
  const int per = CI * 64;
  for (int i = tid; i < 6 * per; i += gridDim.x * blockDim.x) {
    int m = i / per, r = i % per;
    const float* w = ws[m];
    out[i] = pack2(w[2 * r], w[2 * r + 1]);
  }
}

// ---------------------------------------------------------------------------
// Zero the stat accumulators (re-run every call: deterministic under replay).
// ---------------------------------------------------------------------------
__global__ void zero_k(float* __restrict__ p, int n) {
  int i = blockIdx.x * blockDim.x + threadIdx.x;
  if (i < n) p[i] = 0.f;
}

// (sum, sumsq) -> (mu, rstd) in place; biased variance matches jnp.var.
__global__ void finalize_stats(float* __restrict__ st) {
  int i = blockIdx.x * blockDim.x + threadIdx.x;
  if (i < NB * CI) {
    float s1 = st[i * 2], s2 = st[i * 2 + 1];
    float mu = s1 / (float)DVOL;
    float var = s2 / (float)DVOL - mu * mu;
    st[i * 2] = mu;
    st[i * 2 + 1] = rsqrtf(var + EPSV);
  }
}

// ---------------------------------------------------------------------------
// K/V projection (100 MFLOP total, VALU kernel, coalesced over l).
// ---------------------------------------------------------------------------
__global__ void kv_kernel(const float* __restrict__ lang, const float* __restrict__ mask,
                          const float* __restrict__ wk, const float* __restrict__ bk,
                          const float* __restrict__ wv, const float* __restrict__ bv,
                          float* __restrict__ kbuf, float* __restrict__ vbuf) {
  int o = blockIdx.x, mat = blockIdx.y, b = blockIdx.z;
  int l = threadIdx.x;
  const float* w  = mat ? wv : wk;
  const float* bi = mat ? bv : bk;
  const float* lb = lang + (size_t)b * CT * LLEN;
  float acc = 0.f;
  for (int c = 0; c < CT; ++c) acc = fmaf(w[o * CT + c], lb[c * LLEN + l], acc);
  acc = (acc + bi[o]) * mask[b * LLEN + l];
  float* dst = (mat ? vbuf : kbuf) + ((size_t)b * CI + o) * LLEN + l;
  *dst = acc;
}

// ---------------------------------------------------------------------------
// Generic 1x1-conv GEMM: Y[b][o][s] = epi( sum_c W[o][c] * stage(X)[c][s] + b[o] ).
// Workgroup tile 128(M=o) x 128(N=s); 8 wave32s; K=128 in 4 chunks of 32.
// Double-buffered LDS tile [n][k] bf16; next chunk's global loads issued
// before the current chunk's WMMAs (software pipelining).
//   NORMPROD: stage = instnorm(X)*instnorm(X2)   (wpo conv)
//   STATS:    accumulate per-channel sum/sumsq of Y via shfl + atomicAdd
//   EPI 0=none 1=relu 2=tanh*aux (final gate)
// ---------------------------------------------------------------------------
#define EPI_NONE 0
#define EPI_RELU 1
#define EPI_TANH_MUL 2

template <int EPI, bool NORMPROD, bool STATS>
__global__ __launch_bounds__(256) void conv_gemm(
    const float* __restrict__ X, const float* __restrict__ X2,
    const float* __restrict__ statA, const float* __restrict__ statB,
    const uint32_t* __restrict__ Wbf, const float* __restrict__ bias,
    const float* __restrict__ aux, float* __restrict__ Y,
    float* __restrict__ ssum) {
  __shared__ uint32_t XT[2][128 * 16];   // double-buffered [n][k-pair] bf16x2

  const int b  = blockIdx.y;
  const int s0 = blockIdx.x * 128;
  const int t    = threadIdx.x;
  const int wave = t >> 5, lane = t & 31;
  const int ln16 = lane & 15, half = lane >> 4;
  const int n = t & 127, kg2 = t >> 7;    // staging role: column n, k-half kg2

  auto load_chunk = [&](int kc, float* v) {
    const int cbase = kc * 32 + kg2 * 16;
    const float* Xb = X + ((size_t)b * CI + cbase) * DVOL + s0 + n;
#pragma unroll
    for (int j = 0; j < 16; ++j) v[j] = Xb[j * (size_t)DVOL];
    if (NORMPROD) {
      const float* X2b = X2 + ((size_t)b * CI + cbase) * DVOL + s0 + n;
#pragma unroll
      for (int j = 0; j < 16; ++j) {
        int c = cbase + j;
        float a = (v[j] - statA[(b * CI + c) * 2]) * statA[(b * CI + c) * 2 + 1];
        float p = (X2b[j * (size_t)DVOL] - statB[(b * CI + c) * 2]) * statB[(b * CI + c) * 2 + 1];
        v[j] = a * p;
      }
    }
  };
  auto store_chunk = [&](uint32_t* buf, const float* v) {
    uint32_t p[8];
#pragma unroll
    for (int j = 0; j < 8; ++j) p[j] = pack2(v[2 * j], v[2 * j + 1]);
    uint32_t* dst = buf + n * 16 + kg2 * 8;       // 16B aligned
    *(uint4*)(dst)     = *(const uint4*)(p);
    *(uint4*)(dst + 4) = *(const uint4*)(p + 4);
  };

  v8f z8 = {0.f, 0.f, 0.f, 0.f, 0.f, 0.f, 0.f, 0.f};
  v8f acc[8];
#pragma unroll
  for (int i = 0; i < 8; ++i) acc[i] = z8;

  {
    float v0[16];
    load_chunk(0, v0);
    store_chunk(XT[0], v0);
  }

#pragma unroll
  for (int kc = 0; kc < 4; ++kc) {
    float nv[16];
    if (kc < 3) load_chunk(kc + 1, nv);     // issue global loads early
    __syncthreads();

    // A operand (16x32 bf16): weight row r; ISA layout: lanes 0-15 hold
    // K 0-7 / 16-23 (dwords off+0..3 / off+8..11), lanes 16-31 K 8-15 / 24-31.
    const int r = wave * 16 + ln16;
    const uint32_t* Wr = Wbf + r * 64 + kc * 16 + half * 4;
    BfPack A;
    A.q[0] = *(const uint4*)(Wr);
    A.q[1] = *(const uint4*)(Wr + 8);

    const uint32_t* XB = XT[kc & 1];
#pragma unroll
    for (int tt = 0; tt < 8; ++tt) {       // B operand (32x16 bf16) per N tile
      const uint32_t* Xr = XB + (tt * 16 + ln16) * 16 + half * 4;
      BfPack Bv;
      Bv.q[0] = *(const uint4*)(Xr);
      Bv.q[1] = *(const uint4*)(Xr + 8);
      acc[tt] = __builtin_amdgcn_wmma_f32_16x16x32_bf16(
          false, A.v, false, Bv.v, (short)0, acc[tt], false, false);
    }

    if (kc < 3) store_chunk(XT[(kc + 1) & 1], nv);
  }

  // Epilogue: C layout -> VGPR e: lanes0-15 M=e, lanes16-31 M=e+8, N=lane%16.
  float bv8[8], s1[8], s2[8];
#pragma unroll
  for (int e = 0; e < 8; ++e) {
    bv8[e] = bias[wave * 16 + e + 8 * half];
    if (STATS) { s1[e] = 0.f; s2[e] = 0.f; }
  }
#pragma unroll
  for (int tt = 0; tt < 8; ++tt) {
#pragma unroll
    for (int e = 0; e < 8; ++e) {
      int o = wave * 16 + e + 8 * half;
      int s = s0 + tt * 16 + ln16;
      float y = acc[tt][e] + bv8[e];
      if (STATS) { s1[e] += y; s2[e] += y * y; }
      size_t idx = ((size_t)b * CI + o) * DVOL + s;
      if (EPI == EPI_RELU)     y = fmaxf(y, 0.f);
      if (EPI == EPI_TANH_MUL) y = tanhf(y) * aux[idx];
      Y[idx] = y;
    }
  }

  if (STATS) {   // reduce each row over its 16-lane half, one atomic per row
#pragma unroll
    for (int e = 0; e < 8; ++e) {
      s1[e] += __shfl_xor(s1[e], 1, 32);  s2[e] += __shfl_xor(s2[e], 1, 32);
      s1[e] += __shfl_xor(s1[e], 2, 32);  s2[e] += __shfl_xor(s2[e], 2, 32);
      s1[e] += __shfl_xor(s1[e], 4, 32);  s2[e] += __shfl_xor(s2[e], 4, 32);
      s1[e] += __shfl_xor(s1[e], 8, 32);  s2[e] += __shfl_xor(s2[e], 8, 32);
      if (ln16 == 0) {
        int o = wave * 16 + e + 8 * half;
        atomicAdd(&ssum[(b * CI + o) * 2],     s1[e]);
        atomicAdd(&ssum[(b * CI + o) * 2 + 1], s2[e]);
      }
    }
  }
}

// ---------------------------------------------------------------------------
// Fused attention: per (b, h, 128-row strip of q-positions).
// S = instnorm(Q)·K (WMMA), masked softmax over L=128 in registers
// (shfl_xor within 16-lane halves), O = P·Vt (WMMA), written to xattn[b][s][c].
// ---------------------------------------------------------------------------
__global__ __launch_bounds__(256) void attn_kernel(
    const float* __restrict__ rawq, const float* __restrict__ statq,
    const float* __restrict__ kbuf, const float* __restrict__ vbuf,
    const float* __restrict__ mask, float* __restrict__ xattn) {
  __shared__ uint32_t QT[128 * 16];       // [s_local][j-pair]  (A for S-GEMM)
  __shared__ uint32_t KT[128 * 16];       // [l][j-pair]        (B for S-GEMM)
  __shared__ uint32_t VT[32 * 64];        // [j][l-pair]        (B for O-GEMM)
  __shared__ uint32_t PT[8 * 16 * 64];    // per-wave probs [16][l-pair]
  __shared__ float    MK[128];

  const int b = blockIdx.z, h = blockIdx.y, s0 = blockIdx.x * 128;
  const int t = threadIdx.x;
  const int wave = t >> 5, lane = t & 31;
  const int ln16 = lane & 15, half = lane >> 4;

  {   // staging
    const int n = t & 127, kg2 = t >> 7;
    const int c0b = h * HD + kg2 * 16;
    const float* qb = rawq + ((size_t)b * CI + c0b) * DVOL + s0 + n;
    const float* kb = kbuf + ((size_t)b * CI + c0b) * LLEN + n;
#pragma unroll
    for (int j = 0; j < 8; ++j) {
      int c0 = c0b + 2 * j, c1 = c0 + 1;
      float q0 = (qb[(2 * j) * (size_t)DVOL]     - statq[(b * CI + c0) * 2]) * statq[(b * CI + c0) * 2 + 1];
      float q1 = (qb[(2 * j + 1) * (size_t)DVOL] - statq[(b * CI + c1) * 2]) * statq[(b * CI + c1) * 2 + 1];
      QT[n * 16 + kg2 * 8 + j] = pack2(q0, q1);
      KT[n * 16 + kg2 * 8 + j] = pack2(kb[(2 * j) * LLEN], kb[(2 * j + 1) * LLEN]);
    }
    const float* vb = vbuf + ((size_t)b * CI + h * HD) * LLEN;
    for (int i = t; i < 32 * 64; i += 256) {
      int j = i >> 6, lp = i & 63;
      VT[i] = pack2(vb[j * LLEN + 2 * lp], vb[j * LLEN + 2 * lp + 1]);
    }
    if (t < 128) MK[t] = mask[b * LLEN + t];
  }
  __syncthreads();

  v8f z8 = {0.f, 0.f, 0.f, 0.f, 0.f, 0.f, 0.f, 0.f};
  v8f S[8];
  {   // S = Qn x K, single K=32 chunk
    const uint32_t* Qr = QT + (wave * 16 + ln16) * 16 + half * 4;
    BfPack A;
    A.q[0] = *(const uint4*)(Qr);
    A.q[1] = *(const uint4*)(Qr + 8);
#pragma unroll
    for (int tt = 0; tt < 8; ++tt) {
      const uint32_t* Kr = KT + (tt * 16 + ln16) * 16 + half * 4;
      BfPack Bv;
      Bv.q[0] = *(const uint4*)(Kr);
      Bv.q[1] = *(const uint4*)(Kr + 8);
      S[tt] = __builtin_amdgcn_wmma_f32_16x16x32_bf16(
          false, A.v, false, Bv.v, (short)0, z8, false, false);
    }
  }

  // masked softmax over l (row = (e,half); N across lanes within 16-lane half)
  const float scale = 0.17677669529663689f;   // 32^-0.5
  float rmax[8], rsum[8];
#pragma unroll
  for (int e = 0; e < 8; ++e) rmax[e] = -3.0e38f;
#pragma unroll
  for (int tt = 0; tt < 8; ++tt) {
    float addv = 10000.f * MK[tt * 16 + ln16] - 10000.f;
#pragma unroll
    for (int e = 0; e < 8; ++e) {
      float v = S[tt][e] * scale + addv;
      S[tt][e] = v;
      rmax[e] = fmaxf(rmax[e], v);
    }
  }
#pragma unroll
  for (int e = 0; e < 8; ++e) {
    rmax[e] = fmaxf(rmax[e], __shfl_xor(rmax[e], 1, 32));
    rmax[e] = fmaxf(rmax[e], __shfl_xor(rmax[e], 2, 32));
    rmax[e] = fmaxf(rmax[e], __shfl_xor(rmax[e], 4, 32));
    rmax[e] = fmaxf(rmax[e], __shfl_xor(rmax[e], 8, 32));
    rsum[e] = 0.f;
  }
#pragma unroll
  for (int tt = 0; tt < 8; ++tt)
#pragma unroll
    for (int e = 0; e < 8; ++e) {
      float p = __expf(S[tt][e] - rmax[e]);
      S[tt][e] = p;
      rsum[e] += p;
    }
#pragma unroll
  for (int e = 0; e < 8; ++e) {
    rsum[e] += __shfl_xor(rsum[e], 1, 32);
    rsum[e] += __shfl_xor(rsum[e], 2, 32);
    rsum[e] += __shfl_xor(rsum[e], 4, 32);
    rsum[e] += __shfl_xor(rsum[e], 8, 32);
    rsum[e] = 1.f / rsum[e];
  }

  // restage normalized probabilities as bf16 A-rows
  uint16_t* PTh = (uint16_t*)PT;
#pragma unroll
  for (int tt = 0; tt < 8; ++tt)
#pragma unroll
    for (int e = 0; e < 8; ++e) {
      int m = e + 8 * half;
      int l = tt * 16 + ln16;
      PTh[(wave * 16 + m) * 128 + l] = f2bf(S[tt][e] * rsum[e]);
    }
  __syncthreads();

  // O = P x Vt : M=16, N=32 (2 tiles), K=128 (4 chunks)
  v8f O[2] = {z8, z8};
#pragma unroll
  for (int kc = 0; kc < 4; ++kc) {
    const uint32_t* Pr = PT + (wave * 16 + ln16) * 64 + kc * 16 + half * 4;
    BfPack A;
    A.q[0] = *(const uint4*)(Pr);
    A.q[1] = *(const uint4*)(Pr + 8);
#pragma unroll
    for (int nt = 0; nt < 2; ++nt) {
      const uint32_t* Vr = VT + (nt * 16 + ln16) * 64 + kc * 16 + half * 4;
      BfPack Bv;
      Bv.q[0] = *(const uint4*)(Vr);
      Bv.q[1] = *(const uint4*)(Vr + 8);
      O[nt] = __builtin_amdgcn_wmma_f32_16x16x32_bf16(
          false, A.v, false, Bv.v, (short)0, O[nt], false, false);
    }
  }

  // write xattn[b][s][c] (c = h*32 + j); the later reshape to [b][c'][s'] is a
  // pure reinterpretation of this buffer.
#pragma unroll
  for (int nt = 0; nt < 2; ++nt)
#pragma unroll
    for (int e = 0; e < 8; ++e) {
      int s = s0 + wave * 16 + e + 8 * half;
      int c = h * HD + nt * 16 + ln16;
      xattn[((size_t)b * DVOL + s) * CI + c] = O[nt][e];
    }
}

// ---------------------------------------------------------------------------
extern "C" void kernel_launch(void* const* d_in, const int* in_sizes, int n_in,
                              void* d_out, int out_size, void* d_ws, size_t ws_size,
                              hipStream_t stream) {
  const float* img  = (const float*)d_in[0];
  const float* lang = (const float*)d_in[1];
  const float* mask = (const float*)d_in[2];
  const float* wq  = (const float*)d_in[3];  const float* bq  = (const float*)d_in[4];
  const float* wk  = (const float*)d_in[5];  const float* bk  = (const float*)d_in[6];
  const float* wv  = (const float*)d_in[7];  const float* bv  = (const float*)d_in[8];
  const float* wpw = (const float*)d_in[9];  const float* bpw = (const float*)d_in[10];
  const float* wpi = (const float*)d_in[11]; const float* bpi = (const float*)d_in[12];
  const float* wpo = (const float*)d_in[13]; const float* bpo = (const float*)d_in[14];
  const float* wg1 = (const float*)d_in[15]; const float* bg1 = (const float*)d_in[16];
  const float* wg2 = (const float*)d_in[17]; const float* bg2 = (const float*)d_in[18];

  char* ws = (char*)d_ws;
  const size_t S = (size_t)NB * CI * DVOL * sizeof(float);   // 33.5 MB
  float* raw_q  = (float*)(ws + 0 * S);
  float* xattn  = (float*)(ws + 1 * S);
  float* raw_pi = (float*)(ws + 2 * S);
  float* raw_pw = (float*)(ws + 3 * S);
  float* outb   = (float*)(ws + 4 * S);
  float* g1b    = raw_q;                    // raw_q dead after attention
  char* small = ws + 5 * S;
  float*    kbuf  = (float*)small;    small += (size_t)NB * CI * LLEN * 4;
  float*    vbuf  = (float*)small;    small += (size_t)NB * CI * LLEN * 4;
  uint32_t* Wbf   = (uint32_t*)small; small += (size_t)6 * CI * 64 * 4;
  float*    stats = (float*)small;    small += (size_t)3 * NB * CI * 2 * 4;
  float* stq  = stats;
  float* stpi = stats + NB * CI * 2;
  float* stpw = stats + 2 * NB * CI * 2;
  const int WSZ = CI * 64;   // u32 per packed weight matrix
  const int NSTAT = 3 * NB * CI * 2;   // 1536 floats to zero each call

  pack_weights_k<<<24, 256, 0, stream>>>(wq, wpi, wpw, wpo, wg1, wg2, Wbf);
  zero_k<<<(NSTAT + 255) / 256, 256, 0, stream>>>(stats, NSTAT);
  kv_kernel<<<dim3(CI, 2, NB), LLEN, 0, stream>>>(lang, mask, wk, bk, wv, bv, kbuf, vbuf);

  dim3 gg(DVOL / 128, NB);
  conv_gemm<EPI_NONE, false, true><<<gg, 256, 0, stream>>>(
      img, nullptr, nullptr, nullptr, Wbf + 0 * WSZ, bq, nullptr, raw_q, stq);
  finalize_stats<<<1, 256, 0, stream>>>(stq);

  attn_kernel<<<dim3(DVOL / 128, NH, NB), 256, 0, stream>>>(
      raw_q, stq, kbuf, vbuf, mask, xattn);

  conv_gemm<EPI_NONE, false, true><<<gg, 256, 0, stream>>>(
      img, nullptr, nullptr, nullptr, Wbf + 1 * WSZ, bpi, nullptr, raw_pi, stpi);
  finalize_stats<<<1, 256, 0, stream>>>(stpi);

  conv_gemm<EPI_NONE, false, true><<<gg, 256, 0, stream>>>(
      xattn, nullptr, nullptr, nullptr, Wbf + 2 * WSZ, bpw, nullptr, raw_pw, stpw);
  finalize_stats<<<1, 256, 0, stream>>>(stpw);

  conv_gemm<EPI_RELU, true, false><<<gg, 256, 0, stream>>>(
      raw_pi, raw_pw, stpi, stpw, Wbf + 3 * WSZ, bpo, nullptr, outb, nullptr);

  conv_gemm<EPI_RELU, false, false><<<gg, 256, 0, stream>>>(
      outb, nullptr, nullptr, nullptr, Wbf + 4 * WSZ, bg1, nullptr, g1b, nullptr);

  conv_gemm<EPI_TANH_MUL, false, false><<<gg, 256, 0, stream>>>(
      g1b, nullptr, nullptr, nullptr, Wbf + 5 * WSZ, bg2, outb, (float*)d_out, nullptr);
}
